// SpatioTemporalEncoder_25924422598982
// MI455X (gfx1250) — matrix operand order repaired and checked by reading, not compiled, not run on hardware
//
#include <hip/hip_runtime.h>
#include <hip/hip_bf16.h>

#define DEV __device__ __forceinline__

typedef __attribute__((ext_vector_type(16))) __bf16 v16bf;
typedef __attribute__((ext_vector_type(8)))  __bf16 v8bf;
typedef __attribute__((ext_vector_type(8)))  float  v8f;

// ---- model constants -------------------------------------------------------
static constexpr int       B_    = 16;
static constexpr int       T_    = 12;
static constexpr int       N_    = 2048;
static constexpr int       D_    = 128;
static constexpr int       INNER = 256;               // D*EXP
static constexpr long long TOK   = (long long)B_ * T_ * N_;   // 393216 tokens
static constexpr int       SLABS = B_ * T_;           // 192 (b,t) slabs

// ---- WMMA helpers (CDNA5 wave32, 16x16x32 bf16, f32 accum) -----------------
// A fragment: 16(M) x 32(K). Per ISA 7.12.2: lanes 0-15 hold row M=lane with
// K = {0..7, 16..23}; lanes 16-31 hold the same rows with K = {8..15, 24..31}.
DEV v16bf load_fragA(const __bf16* __restrict__ base, long long row0, int ld, int k0) {
  const int lane = threadIdx.x & 31;
  const long long r = row0 + (lane & 15);
  const int kk = k0 + ((lane & 16) ? 8 : 0);
  const __bf16* p = base + r * (long long)ld + kk;
  v8bf lo = *reinterpret_cast<const v8bf*>(p);        // K kk..kk+7
  v8bf hi = *reinterpret_cast<const v8bf*>(p + 16);   // K kk+16..kk+23
  return __builtin_shufflevector(lo, hi, 0,1,2,3,4,5,6,7,8,9,10,11,12,13,14,15);
}

// B fragment: 32(K) x 16(N), read from a transposed (N x K) array.
// Same K order as A: lanes 0-15 hold column N=lane with K = {0..7, 16..23},
// lanes 16-31 hold the same columns with K = {8..15, 24..31}.
DEV v16bf load_fragB(const __bf16* __restrict__ baseT, long long col0, int ld, int k0) {
  const int lane = threadIdx.x & 31;
  const long long c = col0 + (lane & 15);
  const int kk = k0 + ((lane & 16) ? 8 : 0);
  const __bf16* p = baseT + c * (long long)ld + kk;
  v8bf lo = *reinterpret_cast<const v8bf*>(p);        // K kk..kk+7
  v8bf hi = *reinterpret_cast<const v8bf*>(p + 16);   // K kk+16..kk+23
  return __builtin_shufflevector(lo, hi, 0,1,2,3,4,5,6,7,8,9,10,11,12,13,14,15);
}

DEV v8f wmma_bf16(v16bf a, v16bf b, v8f c) {
  return __builtin_amdgcn_wmma_f32_16x16x32_bf16(false, a, false, b, (short)0, c,
                                                 false, false);
}

DEV v8f vzero8() { return (v8f){0.f,0.f,0.f,0.f,0.f,0.f,0.f,0.f}; }

// C/D element (j) coordinates: lanes 0-15 -> (M=j,   N=lane),
//                              lanes16-31 -> (M=j+8, N=lane-16).
DEV int cd_mbase() { return (threadIdx.x & 16) ? 8 : 0; }
DEV int cd_ncol()  { return threadIdx.x & 15; }

// 32(m) x 64(n) wave tile: 2 A-frags, 4 B-frags, 8 accumulators.
// 12 b128 loads feed 8 WMMAs (vs 18 for a 16x128 tile).
struct TileFrag { v16bf a[2]; v16bf b[4]; };

DEV void mma_tile(const TileFrag& f, v8f acc[2][4]) {
#pragma unroll
  for (int ai = 0; ai < 2; ++ai)
#pragma unroll
    for (int nf = 0; nf < 4; ++nf)
      acc[ai][nf] = wmma_bf16(f.a[ai], f.b[nf], acc[ai][nf]);
}

DEV TileFrag load_tile(const __bf16* __restrict__ A, long long m0, int lda,
                       const __bf16* __restrict__ BT, long long c0, int ldb, int k0) {
  TileFrag f;
  f.a[0] = load_fragA(A, m0,      lda, k0);
  f.a[1] = load_fragA(A, m0 + 16, lda, k0);
#pragma unroll
  for (int nf = 0; nf < 4; ++nf)
    f.b[nf] = load_fragB(BT, c0 + nf * 16, ldb, k0);
  return f;
}

// ---- activation helpers ----------------------------------------------------
DEV float gelu_exact(float x)  { return 0.5f * x * (1.f + erff(x * 0.70710678118654752f)); }
DEV float softplus_f(float x)  { return (x > 20.f) ? x : log1pf(expf(x)); }
DEV float sigmoid_f(float x)   { return 1.f / (1.f + expf(-x)); }

// ============================================================================
// Graph diffusion GEMM:  Y[bt][m,d] = sum_n sT[m,n] * X[bt][n,d]
//   sT : 2048 x 2048 bf16 (pre-transposed support)
//   Xt : per-slab 128(d) x 2048(n) bf16  (d-major => contiguous B fragments)
//   Y  : per-slab 2048(m) x 128(d) bf16  (row-major, feeds GCN projection as A)
//   Yt : per-slab 128(d) x 2048(m) bf16  (d-major, feeds next chained GEMM; may be null)
// grid (2048/64, 192), block 128 (4 waves, 2x2 over 64m x 128d).
// K loop is software-pipelined (ping-pong fragments, unroll 2).
// ============================================================================
__global__ __launch_bounds__(128)
void k_gemm_graph(const __bf16* __restrict__ sT, const __bf16* __restrict__ Xt,
                  __bf16* __restrict__ Y, __bf16* __restrict__ Yt) {
  const int bt   = blockIdx.y;
  const int wave = threadIdx.x >> 5;
  const long long m0 = (long long)blockIdx.x * 64 + (wave & 1) * 32;
  const int       d0 = (wave >> 1) * 64;
  const __bf16* XtS = Xt + (long long)bt * ((long long)D_ * N_);

  v8f acc[2][4];
#pragma unroll
  for (int ai = 0; ai < 2; ++ai)
#pragma unroll
    for (int nf = 0; nf < 4; ++nf) acc[ai][nf] = vzero8();

  TileFrag cur = load_tile(sT, m0, N_, XtS, d0, N_, 0);
#pragma unroll 2
  for (int k0 = 32; k0 < N_; k0 += 32) {
    TileFrag nxt = load_tile(sT, m0, N_, XtS, d0, N_, k0);
    mma_tile(cur, acc);
    cur = nxt;
  }
  mma_tile(cur, acc);

  __bf16* YS  = Y + (long long)bt * ((long long)N_ * D_);
  __bf16* YtS = Yt ? Yt + (long long)bt * ((long long)D_ * N_) : nullptr;
  const int mb = cd_mbase(), nl = cd_ncol();
#pragma unroll
  for (int ai = 0; ai < 2; ++ai) {
#pragma unroll
    for (int nf = 0; nf < 4; ++nf) {
      const int d = d0 + nf * 16 + nl;
#pragma unroll
      for (int j = 0; j < 8; ++j) {
        const long long m = m0 + ai * 16 + mb + j;
        const __bf16 v = (__bf16)acc[ai][nf][j];
        YS[m * D_ + d] = v;
        if (YtS) YtS[(long long)d * N_ + m] = v;
      }
    }
  }
}

// ============================================================================
// GCN projection: sp[r,d] = gelu( sum_p pieces[p][r,:] @ W[p*128:,:]  + b[d] )
// pieces = {h, s0h, s0^2h, s1h, s1^2h}; WT is 128(d) x 640(k) bf16.
// grid (TOK/64), block 128 (waves 2x2 over 64r x 128d). 20 k-tiles, pipelined.
// ============================================================================
__global__ __launch_bounds__(128)
void k_gemm_sp(const __bf16* __restrict__ p0, const __bf16* __restrict__ p1,
               const __bf16* __restrict__ p2, const __bf16* __restrict__ p3,
               const __bf16* __restrict__ p4, const __bf16* __restrict__ WT,
               const float* __restrict__ bias, float* __restrict__ sp) {
  const int wave = threadIdx.x >> 5;
  const long long r0 = (long long)blockIdx.x * 64 + (wave & 1) * 32;
  const int       c0 = (wave >> 1) * 64;
  const __bf16* pieces[5] = {p0, p1, p2, p3, p4};

  v8f acc[2][4];
#pragma unroll
  for (int ai = 0; ai < 2; ++ai)
#pragma unroll
    for (int nf = 0; nf < 4; ++nf) acc[ai][nf] = vzero8();

  TileFrag cur = load_tile(pieces[0], r0, 128, WT, c0, 640, 0);
#pragma unroll
  for (int kt = 1; kt < 20; ++kt) {
    TileFrag nxt = load_tile(pieces[kt >> 2], r0, 128, WT, c0, 640, 0);
    // patch the A/B k-offsets: A uses local k, B uses global k
    nxt.a[0] = load_fragA(pieces[kt >> 2], r0,      128, (kt & 3) * 32);
    nxt.a[1] = load_fragA(pieces[kt >> 2], r0 + 16, 128, (kt & 3) * 32);
#pragma unroll
    for (int nf = 0; nf < 4; ++nf)
      nxt.b[nf] = load_fragB(WT, c0 + nf * 16, 640, kt * 32);
    mma_tile(cur, acc);
    cur = nxt;
  }
  mma_tile(cur, acc);

  const int mb = cd_mbase(), nl = cd_ncol();
#pragma unroll
  for (int ai = 0; ai < 2; ++ai) {
#pragma unroll
    for (int nf = 0; nf < 4; ++nf) {
      const int d = c0 + nf * 16 + nl;
#pragma unroll
      for (int j = 0; j < 8; ++j) {
        const long long r = r0 + ai * 16 + mb + j;
        sp[r * D_ + d] = gelu_exact(acc[ai][nf][j] + bias[d]);
      }
    }
  }
}

// ============================================================================
// Generic GEMM: C = act( A(bf16, MxK) @ BT(bf16, NxK)^T + bias )
//   act: 0=none, 1=softplus, 2=tanh
//   mode 0: Cout[r*Ncols + c]  (rows in whatever order A rows are)
//   mode 1: "pr" scatter: rows are (b,t,n); c<256 -> sig, c>=256 -> gate,
//           both written in (B,N,T,256) order for the conv/scan stage.
// grid (M/64, Ncols/128), block 128 (waves 2x2). Pipelined K loop.
// ============================================================================
__global__ __launch_bounds__(128)
void k_gemm_gen(const __bf16* __restrict__ A, int K, const __bf16* __restrict__ BT,
                const float* __restrict__ bias, float* __restrict__ Cout, int Ncols,
                int act, int mode, float* __restrict__ sigOut,
                float* __restrict__ gateOut) {
  const int wave = threadIdx.x >> 5;
  const long long r0 = (long long)blockIdx.x * 64 + (wave & 1) * 32;
  const int       c0 = blockIdx.y * 128 + (wave >> 1) * 64;

  v8f acc[2][4];
#pragma unroll
  for (int ai = 0; ai < 2; ++ai)
#pragma unroll
    for (int nf = 0; nf < 4; ++nf) acc[ai][nf] = vzero8();

  TileFrag cur = load_tile(A, r0, K, BT, c0, K, 0);
#pragma unroll 2
  for (int k0 = 32; k0 < K; k0 += 32) {
    TileFrag nxt = load_tile(A, r0, K, BT, c0, K, k0);
    mma_tile(cur, acc);
    cur = nxt;
  }
  mma_tile(cur, acc);

  const int mb = cd_mbase(), nl = cd_ncol();
#pragma unroll
  for (int ai = 0; ai < 2; ++ai) {
#pragma unroll
    for (int nf = 0; nf < 4; ++nf) {
      const int c = c0 + nf * 16 + nl;
#pragma unroll
      for (int j = 0; j < 8; ++j) {
        const long long r = r0 + ai * 16 + mb + j;
        float x = acc[ai][nf][j] + bias[c];
        if (act == 1)      x = softplus_f(x);
        else if (act == 2) x = tanhf(x);
        if (mode == 0) {
          Cout[r * (long long)Ncols + c] = x;
        } else {
          const long long b = r / ((long long)T_ * N_);
          const int t = (int)((r / N_) % T_);
          const int n = (int)(r % N_);
          const long long row2 = (b * N_ + n) * T_ + t;
          if (c < INNER) sigOut [row2 * INNER + c]           = x;
          else           gateOut[row2 * INNER + (c - INNER)] = x;
        }
      }
    }
  }
}

// ---- elementwise / small kernels -------------------------------------------
__global__ void k_trans_support(const float* __restrict__ s, __bf16* __restrict__ sT) {
  const long long idx = (long long)blockIdx.x * 256 + threadIdx.x;
  const long long per = (long long)N_ * N_;
  if (idx >= 2 * per) return;
  const long long sidx = idx / per, rem = idx % per;
  const int k = (int)(rem / N_), m = (int)(rem % N_);
  sT[sidx * per + (long long)m * N_ + k] = (__bf16)s[idx];
}

__global__ void k_convert_h(const float* __restrict__ h, __bf16* __restrict__ hb,
                            __bf16* __restrict__ hbT) {
  const long long idx = (long long)blockIdx.x * 256 + threadIdx.x;
  if (idx >= TOK * D_) return;
  const __bf16 bv = (__bf16)h[idx];
  hb[idx] = bv;
  const long long r = idx >> 7;        // token (b,t,n)
  const int d  = (int)(idx & 127);
  const long long bt = r >> 11;        // slab
  const int n  = (int)(r & 2047);
  hbT[bt * ((long long)D_ * N_) + (long long)d * N_ + n] = bv;
}

__global__ void k_trans_w(const float* __restrict__ W, __bf16* __restrict__ WT,
                          int K, int N) {
  const int idx = blockIdx.x * 256 + threadIdx.x;
  if (idx >= K * N) return;
  const int k = idx / N, n = idx % N;
  WT[n * K + k] = (__bf16)W[idx];
}

// causal depthwise conv (K=4) + SiLU ; sig (B,N,T,256) f32 -> sig2 bf16
__global__ void k_conv_silu(const float* __restrict__ sig, const float* __restrict__ cw,
                            const float* __restrict__ cb, __bf16* __restrict__ sig2) {
  const long long idx = (long long)blockIdx.x * 256 + threadIdx.x;
  if (idx >= (long long)B_ * N_ * INNER) return;
  const long long bn = idx >> 8;
  const int c = (int)(idx & 255);
  const float w0 = cw[c*4+0], w1 = cw[c*4+1], w2 = cw[c*4+2], w3 = cw[c*4+3];
  const float bias = cb[c];
  const float* srow = sig  + bn * (long long)T_ * INNER + c;
  __bf16*      drow = sig2 + bn * (long long)T_ * INNER + c;
  float x0 = 0.f, x1 = 0.f, x2 = 0.f;
#pragma unroll
  for (int t = 0; t < T_; ++t) {
    const float x3 = srow[t * INNER];
    const float a = bias + w0*x0 + w1*x1 + w2*x2 + w3*x3;
    drow[t * INNER] = (__bf16)(a * sigmoid_f(a));
    x0 = x1; x1 = x2; x2 = x3;
  }
}

// selective scan over T=12 + gate ; outputs hid bf16 in (B,N,T,256)
__global__ void k_scan(const __bf16* __restrict__ sig2, const float* __restrict__ delta,
                       const float* __restrict__ s_in, const float* __restrict__ s_out,
                       const float* __restrict__ gate, const float* __restrict__ a_log,
                       const float* __restrict__ d_skip, __bf16* __restrict__ hid) {
  const long long idx = (long long)blockIdx.x * 256 + threadIdx.x;
  if (idx >= (long long)B_ * N_ * INNER) return;
  const long long bn = idx >> 8;
  const int c = (int)(idx & 255);
  const float a = -softplus_f(a_log[c]);
  const float d = d_skip[c];
  const long long base = bn * (long long)T_ * INNER + c;
  float state = 0.f;
#pragma unroll
  for (int t = 0; t < T_; ++t) {
    const long long o = base + (long long)t * INNER;
    const float x  = (float)sig2[o];
    const float dl = delta[o];
    state = expf(dl * a) * state + s_in[o] * x;
    const float y = s_out[o] * state + d * x;
    hid[o] = (__bf16)(y * sigmoid_f(gate[o]));
  }
}

// fused residual combine + LayerNorm; one wave per token.
__global__ __launch_bounds__(128)
void k_fuse_ln(const float* __restrict__ h, const float* __restrict__ sp,
               const float* __restrict__ tp, const float* __restrict__ g,
               const float* __restrict__ bb, float* __restrict__ out) {
  const int wave = threadIdx.x >> 5, lane = threadIdx.x & 31;
  const long long r = (long long)blockIdx.x * 4 + wave;      // token (b,t,n)
  const long long b = r / ((long long)T_ * N_);
  const int t = (int)((r / N_) % T_);
  const int n = (int)(r % N_);
  const long long tpr = ((b * N_ + n) * T_ + t) * (long long)D_;

  const float4 hv = reinterpret_cast<const float4*>(h  + r  * D_)[lane];
  const float4 sv = reinterpret_cast<const float4*>(sp + r  * D_)[lane];
  const float4 tv = reinterpret_cast<const float4*>(tp + tpr)[lane];
  float f[4];
  f[0] = hv.x + sv.x + tv.x + sv.x * tv.x;
  f[1] = hv.y + sv.y + tv.y + sv.y * tv.y;
  f[2] = hv.z + sv.z + tv.z + sv.z * tv.z;
  f[3] = hv.w + sv.w + tv.w + sv.w * tv.w;

  float sum = f[0] + f[1] + f[2] + f[3];
  float sq  = f[0]*f[0] + f[1]*f[1] + f[2]*f[2] + f[3]*f[3];
#pragma unroll
  for (int m = 16; m > 0; m >>= 1) {
    sum += __shfl_xor(sum, m);
    sq  += __shfl_xor(sq,  m);
  }
  const float mu  = sum * (1.f / D_);
  const float var = sq * (1.f / D_) - mu * mu;
  const float inv = rsqrtf(var + 1e-5f);

  float4 o;
  const int d0 = lane * 4;
  o.x = (f[0] - mu) * inv * g[d0+0] + bb[d0+0];
  o.y = (f[1] - mu) * inv * g[d0+1] + bb[d0+1];
  o.z = (f[2] - mu) * inv * g[d0+2] + bb[d0+2];
  o.w = (f[3] - mu) * inv * g[d0+3] + bb[d0+3];
  reinterpret_cast<float4*>(out + r * D_)[lane] = o;
}

// ============================================================================
// host orchestration
// ============================================================================
extern "C" void kernel_launch(void* const* d_in, const int* in_sizes, int n_in,
                              void* d_out, int out_size, void* d_ws, size_t ws_size,
                              hipStream_t stream) {
  (void)in_sizes; (void)n_in; (void)out_size; (void)ws_size;

  const float* inputs   = (const float*)d_in[0];
  const float* supports = (const float*)d_in[1];
  const float* gcn_w    = (const float*)d_in[2];
  const float* gcn_b    = (const float*)d_in[3];
  const float* in_w     = (const float*)d_in[4];
  const float* in_b     = (const float*)d_in[5];
  const float* conv_w   = (const float*)d_in[6];
  const float* conv_b   = (const float*)d_in[7];
  const float* delta_w  = (const float*)d_in[8];
  const float* delta_b  = (const float*)d_in[9];
  const float* si_w     = (const float*)d_in[10];
  const float* si_b     = (const float*)d_in[11];
  const float* so_w     = (const float*)d_in[12];
  const float* so_b     = (const float*)d_in[13];
  const float* a_log    = (const float*)d_in[14];
  const float* d_skip   = (const float*)d_in[15];
  const float* out_w    = (const float*)d_in[16];
  const float* out_b    = (const float*)d_in[17];
  const float* ln_g     = (const float*)d_in[18];
  const float* ln_b     = (const float*)d_in[19];
  float* outp = (float*)d_out;

  char* ws = (char*)d_ws;
  size_t off = 0;
  auto alloc = [&](size_t bytes) -> void* {
    void* p = ws + off;
    off += (bytes + 255) & ~(size_t)255;
    return p;
  };

  const size_t tokD_bf  = (size_t)TOK * D_ * sizeof(__bf16);
  const size_t tokD_f   = (size_t)TOK * D_ * sizeof(float);
  const size_t tokI_bf  = (size_t)TOK * INNER * sizeof(__bf16);
  const size_t tokI_f   = (size_t)TOK * INNER * sizeof(float);

  __bf16* sT0   = (__bf16*)alloc(2 * (size_t)N_ * N_ * sizeof(__bf16));
  __bf16* sT1   = sT0 + (size_t)N_ * N_;
  float*  hbuf  = (float*)alloc(tokD_f);
  __bf16* h_bf  = (__bf16*)alloc(tokD_bf);
  __bf16* h_bfT = (__bf16*)alloc(tokD_bf);
  __bf16* fbase = (__bf16*)alloc(4 * tokD_bf);          // f1..f4 ; reused as delta
  __bf16* f1    = fbase, *f2 = fbase + TOK*D_, *f3 = fbase + 2*TOK*D_, *f4 = fbase + 3*TOK*D_;
  __bf16* f1T   = (__bf16*)alloc(tokD_bf);
  __bf16* f3T   = (__bf16*)alloc(tokD_bf);
  float*  sp    = (float*)alloc(tokD_f);
  float*  sig   = (float*)alloc(tokI_f);                // reused as s_in
  float*  gate  = (float*)alloc(tokI_f);
  float*  so_b_ = (float*)alloc(tokI_f);
  __bf16* sig2  = (__bf16*)alloc(tokI_bf);
  __bf16* hid   = (__bf16*)alloc(tokI_bf);
  float*  tp    = (float*)alloc(tokD_f);
  __bf16* gcn_wT = (__bf16*)alloc((size_t)128 * 640 * sizeof(__bf16));
  __bf16* in_wT  = (__bf16*)alloc((size_t)512 * 128 * sizeof(__bf16));
  __bf16* dwT    = (__bf16*)alloc((size_t)256 * 256 * sizeof(__bf16));
  __bf16* siwT   = (__bf16*)alloc((size_t)256 * 256 * sizeof(__bf16));
  __bf16* sowT   = (__bf16*)alloc((size_t)256 * 256 * sizeof(__bf16));
  __bf16* owT    = (__bf16*)alloc((size_t)128 * 256 * sizeof(__bf16));

  float* delta_buf = (float*)fbase;   // reuse: f's dead after GCN projection
  float* si_buf    = sig;             // reuse: sig dead after conv

  // one-time: transpose+convert supports
  k_trans_support<<<(2LL * N_ * N_ + 255) / 256, 256, 0, stream>>>(supports, sT0);

  const dim3 gGraph(N_ / 64, SLABS);
  const int  gRows = (int)(TOK / 64);                   // 6144

  for (int l = 0; l < 2; ++l) {
    const float* h_src = (l == 0) ? inputs : hbuf;
    float*       h_dst = (l == 1) ? outp   : hbuf;

    const float* gcn_w_l   = gcn_w   + (size_t)l * 640 * 128;
    const float* gcn_b_l   = gcn_b   + (size_t)l * 128;
    const float* in_w_l    = in_w    + (size_t)l * 128 * 512;
    const float* in_b_l    = in_b    + (size_t)l * 512;
    const float* conv_w_l  = conv_w  + (size_t)l * 256 * 4;
    const float* conv_b_l  = conv_b  + (size_t)l * 256;
    const float* delta_w_l = delta_w + (size_t)l * 256 * 256;
    const float* delta_b_l = delta_b + (size_t)l * 256;
    const float* si_w_l    = si_w    + (size_t)l * 256 * 256;
    const float* si_b_l    = si_b    + (size_t)l * 256;
    const float* so_w_l    = so_w    + (size_t)l * 256 * 256;
    const float* so_b_l    = so_b    + (size_t)l * 256;
    const float* a_log_l   = a_log   + (size_t)l * 256;
    const float* d_skip_l  = d_skip  + (size_t)l * 256;
    const float* out_w_l   = out_w   + (size_t)l * 256 * 128;
    const float* out_b_l   = out_b   + (size_t)l * 128;
    const float* ln_g_l    = ln_g    + (size_t)l * 128;
    const float* ln_b_l    = ln_b    + (size_t)l * 128;

    // weight convert+transpose (tiny)
    k_trans_w<<<(640*128 + 255)/256, 256, 0, stream>>>(gcn_w_l,   gcn_wT, 640, 128);
    k_trans_w<<<(128*512 + 255)/256, 256, 0, stream>>>(in_w_l,    in_wT,  128, 512);
    k_trans_w<<<(256*256 + 255)/256, 256, 0, stream>>>(delta_w_l, dwT,    256, 256);
    k_trans_w<<<(256*256 + 255)/256, 256, 0, stream>>>(si_w_l,    siwT,   256, 256);
    k_trans_w<<<(256*256 + 255)/256, 256, 0, stream>>>(so_w_l,    sowT,   256, 256);
    k_trans_w<<<(256*128 + 255)/256, 256, 0, stream>>>(out_w_l,   owT,    256, 128);

    // h -> bf16 (row-major + d-major)
    k_convert_h<<<(TOK * D_ + 255) / 256, 256, 0, stream>>>(h_src, h_bf, h_bfT);

    // graph diffusion chain (the 1.65 TFLOP hot path)
    k_gemm_graph<<<gGraph, 128, 0, stream>>>(sT0, h_bfT, f1, f1T);
    k_gemm_graph<<<gGraph, 128, 0, stream>>>(sT0, f1T,   f2, nullptr);
    k_gemm_graph<<<gGraph, 128, 0, stream>>>(sT1, h_bfT, f3, f3T);
    k_gemm_graph<<<gGraph, 128, 0, stream>>>(sT1, f3T,   f4, nullptr);

    // GCN projection + exact GELU
    k_gemm_sp<<<gRows, 128, 0, stream>>>(h_bf, f1, f2, f3, f4, gcn_wT, gcn_b_l, sp);

    // temporal input projection -> signal/gate in (B,N,T,256)
    k_gemm_gen<<<dim3(gRows, 4), 128, 0, stream>>>(h_bf, 128, in_wT, in_b_l,
                                                   nullptr, 512, 0, 1, sig, gate);

    // causal depthwise conv + SiLU
    k_conv_silu<<<((long long)B_*N_*INNER + 255)/256, 256, 0, stream>>>(
        sig, conv_w_l, conv_b_l, sig2);

    // delta / s_in / s_out projections
    k_gemm_gen<<<dim3(gRows, 2), 128, 0, stream>>>(sig2, 256, dwT,  delta_b_l,
                                                   delta_buf, 256, 1, 0, nullptr, nullptr);
    k_gemm_gen<<<dim3(gRows, 2), 128, 0, stream>>>(sig2, 256, siwT, si_b_l,
                                                   si_buf,    256, 2, 0, nullptr, nullptr);
    k_gemm_gen<<<dim3(gRows, 2), 128, 0, stream>>>(sig2, 256, sowT, so_b_l,
                                                   so_b_,     256, 2, 0, nullptr, nullptr);

    // selective scan + gate
    k_scan<<<((long long)B_*N_*INNER + 255)/256, 256, 0, stream>>>(
        sig2, delta_buf, si_buf, so_b_, gate, a_log_l, d_skip_l, hid);

    // output projection
    k_gemm_gen<<<dim3(gRows, 1), 128, 0, stream>>>(hid, 256, owT, out_b_l,
                                                   tp, 128, 0, 0, nullptr, nullptr);

    // fuse + LayerNorm
    k_fuse_ln<<<(int)(TOK / 4), 128, 0, stream>>>(h_src, sp, tp, ln_g_l, ln_b_l, h_dst);
  }
}
